// Encoder_5325759447269
// MI455X (gfx1250) — matrix operand order repaired and checked by reading, not compiled, not run on hardware
//
#include <hip/hip_runtime.h>
#include <hip/hip_bf16.h>
#include <stdint.h>
#include <stddef.h>

// ---------------------------------------------------------------------------
// CDNA5 (gfx1250) CRNN encoder.
// Conv stack: implicit-GEMM WMMA f16, NHWC activations with zero-padded halos,
//   K ordered (tap, channel) so B-fragments are contiguous b128 loads.
// LSTM: input projection hoisted to two big WMMA GEMMs; per-step recurrent
//   GEMM fused with the gate nonlinearity + cell update (wave32, f32 accum).
// ---------------------------------------------------------------------------

typedef __attribute__((ext_vector_type(16))) _Float16 v16h;
typedef __attribute__((ext_vector_type(8)))  _Float16 v8h;
typedef __attribute__((ext_vector_type(8)))  float    v8f;

#define DEVFN static __device__ __forceinline__

DEVFN v8f wmma16(v16h a, v16h b, v8f c) {
  return __builtin_amdgcn_wmma_f32_16x16x32_f16(false, a, false, b, (short)0, c,
                                                false, false);
}

DEVFN v8f vzero8() {
  v8f z;
#pragma unroll
  for (int i = 0; i < 8; ++i) z[i] = 0.f;
  return z;
}

// A-fragment (16x32 f16, row-major, leading dim lda).
// Lane L: row m = L&15; half g = L>>4 holds K = 16g..16g+15.
DEVFN v16h load_fragA(const _Float16* __restrict__ tile, int lda, int lane) {
  int row = lane & 15;
  int g   = lane >> 4;
  const _Float16* p = tile + (size_t)row * lda + 16 * g;
  v8h lo = *(const v8h*)(p);
  v8h hi = *(const v8h*)(p + 8);
  v16h r;
#pragma unroll
  for (int i = 0; i < 8; ++i) { r[i] = lo[i]; r[i + 8] = hi[i]; }
  return r;
}

// B-fragment (32x16 f16) with k contiguous at the given per-lane pointer.
// Lane L: column n = L&15 (caller bakes n into p); half g holds K = 16g..16g+15.
DEVFN v16h load_fragB_ptr(const _Float16* __restrict__ p) {
  v8h lo = *(const v8h*)(p);
  v8h hi = *(const v8h*)(p + 8);
  v16h r;
#pragma unroll
  for (int i = 0; i < 8; ++i) { r[i] = lo[i]; r[i + 8] = hi[i]; }
  return r;
}

DEVFN float fast_sig(float x) { return 1.f / (1.f + __expf(-x)); }
DEVFN float fast_tanh(float x) { return 2.f / (1.f + __expf(-2.f * x)) - 1.f; }

// ---------------------------------------------------------------------------
// Prep kernels
// ---------------------------------------------------------------------------

__global__ void prep_x0(const float* __restrict__ src, _Float16* __restrict__ x0, int n) {
  int i = blockIdx.x * blockDim.x + threadIdx.x;
  if (i < n) x0[i] = (_Float16)(src[i] - 0.5f);
}

// Fold optional BN into conv weights/bias, permute K to (tap, channel) order,
// emit f16 weights [Cout x 9*Cin].
__global__ void fold_conv_w(const float* __restrict__ w, const float* __restrict__ b,
                            const float* __restrict__ bg, const float* __restrict__ bb,
                            const float* __restrict__ bm, const float* __restrict__ bv,
                            _Float16* __restrict__ wh, float* __restrict__ bias,
                            int Cout, int Cin) {
  int K = Cin * 9;
  int i = blockIdx.x * blockDim.x + threadIdx.x;
  if (i >= Cout * K) return;
  int co = i / K;
  int k  = i - co * K;
  int rs = k / Cin;
  int ci = k - rs * Cin;
  float scale = 1.f;
  if (bg) scale = bg[co] * rsqrtf(bv[co] + 1e-5f);
  wh[i] = (_Float16)(w[((size_t)co * Cin + ci) * 9 + rs] * scale);
  if (k == 0) {
    float bo = b[co];
    bias[co] = bg ? (bo - bm[co]) * scale + bb[co] : bo;
  }
}

__global__ void cvt_f16(const float* __restrict__ a, _Float16* __restrict__ o, int n) {
  int i = blockIdx.x * blockDim.x + threadIdx.x;
  if (i < n) o[i] = (_Float16)a[i];
}

__global__ void add_bias2(const float* __restrict__ a, const float* __restrict__ b,
                          float* __restrict__ o, int n) {
  int i = blockIdx.x * blockDim.x + threadIdx.x;
  if (i < n) o[i] = a[i] + b[i];
}

__global__ void zero_f32(float* __restrict__ p, int n) {
  int i = blockIdx.x * blockDim.x + threadIdx.x;
  if (i < n) p[i] = 0.f;
}

__global__ void zero_f16(_Float16* __restrict__ p, int n) {
  int i = blockIdx.x * blockDim.x + threadIdx.x;
  if (i < n) p[i] = (_Float16)0.f;
}

// Zero the 1-pixel halo border of an NHWC padded tensor [Bn, Hp2, Wp2, C].
__global__ void zero_border(_Float16* __restrict__ buf, int Bn, int Hp2, int Wp2, int C) {
  int tot = Bn * Hp2 * Wp2 * C;
  int idx = blockIdx.x * blockDim.x + threadIdx.x;
  if (idx >= tot) return;
  int t = idx / C;
  int x = t % Wp2; t /= Wp2;
  int y = t % Hp2;
  if (x == 0 || x == Wp2 - 1 || y == 0 || y == Hp2 - 1) buf[idx] = (_Float16)0.f;
}

// ---------------------------------------------------------------------------
// conv1 (Cin=1) fused relu + 2x2 maxpool (scalar; ~0.4% of conv FLOPs)
// x0: [32, 64, 512] f16  ->  p1 NHWC padded: [32, 34, 258, 64]
// ---------------------------------------------------------------------------
__global__ void conv1_pool(const _Float16* __restrict__ x0,
                           const float* __restrict__ w1, const float* __restrict__ b1,
                           _Float16* __restrict__ out) {
  int idx = blockIdx.x * blockDim.x + threadIdx.x;
  const int TOT = 32 * 32 * 256 * 64;
  if (idx >= TOT) return;
  int co = idx & 63; int t = idx >> 6;
  int x2 = t & 255; t >>= 8;
  int y2 = t & 31; int b = t >> 5;
  float wl[9];
#pragma unroll
  for (int i = 0; i < 9; ++i) wl[i] = w1[co * 9 + i];
  float bias = b1[co];
  float best = 0.f;  // relu floor
#pragma unroll
  for (int dy = 0; dy < 2; ++dy)
#pragma unroll
    for (int dx = 0; dx < 2; ++dx) {
      int y = 2 * y2 + dy, x = 2 * x2 + dx;
      float s = bias;
#pragma unroll
      for (int r = 0; r < 3; ++r) {
        int yy = y + r - 1;
        if (yy < 0 || yy >= 64) continue;
#pragma unroll
        for (int ss = 0; ss < 3; ++ss) {
          int xx = x + ss - 1;
          if (xx < 0 || xx >= 512) continue;
          s += wl[r * 3 + ss] * (float)x0[((size_t)b * 64 + yy) * 512 + xx];
        }
      }
      if (s > best) best = s;
    }
  out[(((size_t)b * 34 + (y2 + 1)) * 258 + (x2 + 1)) * 64 + co] = (_Float16)best;
}

// ---------------------------------------------------------------------------
// 3x3/pad1 conv as implicit GEMM. NHWC in/out with zero halos, so the inner
// loop is pure b128 loads + WMMA (no bounds checks, EXEC untouched).
// One wave -> 32(Cout) x 16(pixel) tile: 2 accumulators share each B fragment.
// in:  [Bn, Hp2, Wp2, Cin] f16   wh: [Cout, 9*Cin] f16 (tap-major K)
// out: [Bn, Hp2, Wp2, Cout] f16 (ReLU, interior only)
// ---------------------------------------------------------------------------
__global__ void conv3x3_wmma(const _Float16* __restrict__ in,
                             const _Float16* __restrict__ wh,
                             const float* __restrict__ bias,
                             _Float16* __restrict__ out,
                             int Bn, int Cin, int Hp2, int Wp2, int Cout) {
  const int H = Hp2 - 2, W = Wp2 - 2;
  const int K = Cin * 9;
  const int HW = H * W;
  const int Ntot = Bn * HW;
  const int Nt = Ntot >> 4;
  const int Mt = Cout >> 5;
  int wid = blockIdx.x * (blockDim.x >> 5) + (threadIdx.x >> 5);
  if (wid >= Mt * Nt) return;
  int mt = wid / Nt, nt = wid - mt * Nt;
  int lane = threadIdx.x & 31;
  int g = lane >> 4;

  int p = nt * 16 + (lane & 15);
  int b = p / HW; int rem = p - b * HW;
  int y = rem / W; int x = rem - y * W;

  // Padded coords: tap (rr,ss) reads (y+rr, x+ss) of the padded tensor.
  const _Float16* pix = in + (((size_t)b * Hp2 + y) * Wp2 + x) * Cin + 16 * g;

  v8f acc0 = vzero8(), acc1 = vzero8();
  const _Float16* A0 = wh + (size_t)(mt * 32) * K;
  const _Float16* A1 = A0 + (size_t)16 * K;

#pragma unroll
  for (int rr = 0; rr < 3; ++rr) {
#pragma unroll
    for (int ss = 0; ss < 3; ++ss) {
      const _Float16* bp = pix + ((size_t)rr * Wp2 + ss) * Cin;
      const int kofs = (rr * 3 + ss) * Cin;
      for (int cb = 0; cb < Cin; cb += 32) {
        v16h bf = load_fragB_ptr(bp + cb);
        v16h a0 = load_fragA(A0 + kofs + cb, K, lane);
        v16h a1 = load_fragA(A1 + kofs + cb, K, lane);
        acc0 = wmma16(a0, bf, acc0);
        acc1 = wmma16(a1, bf, acc1);
      }
    }
  }

  int half = lane >> 4;
  _Float16* op = out + (((size_t)b * Hp2 + (y + 1)) * Wp2 + (x + 1)) * Cout + mt * 32;
  v8h o0, o1;
#pragma unroll
  for (int r = 0; r < 8; ++r) {
    int m0 = mt * 32 + half * 8 + r;
    float v0 = acc0[r] + bias[m0];
    float v1 = acc1[r] + bias[m0 + 16];
    o0[r] = (_Float16)(v0 > 0.f ? v0 : 0.f);
    o1[r] = (_Float16)(v1 > 0.f ? v1 : 0.f);
  }
  *(v8h*)(op + half * 8)      = o0;
  *(v8h*)(op + 16 + half * 8) = o1;
}

// ---------------------------------------------------------------------------
// Max pool, NHWC padded in/out (channel-fastest -> coalesced).
// ---------------------------------------------------------------------------
__global__ void maxpool_f16(const _Float16* __restrict__ in, _Float16* __restrict__ out,
                            int Bn, int C, int H, int W, int kh, int kw) {
  int Ho = H / kh, Wo = W / kw;
  int Hp2 = H + 2, Wp2 = W + 2;
  int Hop2 = Ho + 2, Wop2 = Wo + 2;
  int tot = Bn * Ho * Wo * C;
  int idx = blockIdx.x * blockDim.x + threadIdx.x;
  if (idx >= tot) return;
  int c = idx % C; int t = idx / C;
  int xo = t % Wo; t /= Wo;
  int yo = t % Ho; int b = t / Ho;
  float best = -1e30f;
  for (int dy = 0; dy < kh; ++dy)
    for (int dx = 0; dx < kw; ++dx) {
      float v = (float)in[(((size_t)b * Hp2 + yo * kh + dy + 1) * Wp2 + xo * kw + dx + 1) * C + c];
      if (v > best) best = v;
    }
  out[(((size_t)b * Hop2 + yo + 1) * Wop2 + xo + 1) * C + c] = (_Float16)best;
}

// ---------------------------------------------------------------------------
// Build LSTM input sequence: seq [65, 256, 512] f16
//   t==0 -> pos[hp, c] ; t>0 -> c6 NHWC padded [32, 10, 66, 512] at (b, hp, t-1)
// ---------------------------------------------------------------------------
__global__ void build_seq(const _Float16* __restrict__ c6, const float* __restrict__ pos,
                          _Float16* __restrict__ seq) {
  const int TOT = 65 * 256 * 512;
  int idx = blockIdx.x * blockDim.x + threadIdx.x;
  if (idx >= TOT) return;
  int c = idx & 511; int t2 = idx >> 9;
  int row = t2 & 255; int t = t2 >> 8;
  int hp = row >> 5, b = row & 31;
  float v = (t == 0) ? pos[hp * 512 + c]
                     : (float)c6[(((size_t)b * 10 + hp + 1) * 66 + t) * 512 + c];
  seq[idx] = (_Float16)v;
}

// ---------------------------------------------------------------------------
// out[M x N] = A[M x K](f16) * Wg[N x K]^T (f16) + bsum[N]
// One wave -> 16 x 32 tile (2 N-tiles share each A fragment).
// ---------------------------------------------------------------------------
__global__ void gemm_xw(const _Float16* __restrict__ A, const _Float16* __restrict__ Wg,
                        const float* __restrict__ bsum, float* __restrict__ out,
                        int M, int N, int K) {
  int Mt = M >> 4, Nt2 = N >> 5;
  int wid = blockIdx.x * (blockDim.x >> 5) + (threadIdx.x >> 5);
  if (wid >= Mt * Nt2) return;
  int mt = wid / Nt2, nt = wid - mt * Nt2;
  int lane = threadIdx.x & 31;
  v8f acc0 = vzero8(), acc1 = vzero8();
  const _Float16* Abase = A + (size_t)(mt * 16) * K;
  const _Float16* B0 = Wg + (size_t)(nt * 32 + (lane & 15)) * K + 16 * (lane >> 4);
  const _Float16* B1 = B0 + (size_t)16 * K;
  for (int k0 = 0; k0 < K; k0 += 32) {
    v16h a = load_fragA(Abase + k0, K, lane);
    acc0 = wmma16(a, load_fragB_ptr(B0 + k0), acc0);
    acc1 = wmma16(a, load_fragB_ptr(B1 + k0), acc1);
  }
  int half = lane >> 4, n = lane & 15;
  int col0 = nt * 32 + n;
  float bb0 = bsum[col0], bb1 = bsum[col0 + 16];
#pragma unroll
  for (int r = 0; r < 8; ++r) {
    int row = mt * 16 + half * 8 + r;
    out[(size_t)row * N + col0]      = acc0[r] + bb0;
    out[(size_t)row * N + col0 + 16] = acc1[r] + bb1;
  }
}

// ---------------------------------------------------------------------------
// One LSTM timestep: gates = xw[t] + h_prev @ Whh^T ; fused cell update.
// One wave owns a 16(batch)x16(hidden) tile -> 4 WMMA chains (i,f,g,o).
// h double-buffered in f16 (hin -> hout). 256 waves.
// ---------------------------------------------------------------------------
__global__ void lstm_step(const _Float16* __restrict__ hin,   // [256 x 256] f16
                          _Float16* __restrict__ hout,        // [256 x 256] f16
                          const _Float16* __restrict__ Whh,   // [1024 x 256] f16
                          const float* __restrict__ xw,       // [T*256 x 1024] f32
                          float* __restrict__ cbuf,           // [256 x 256] f32 (in place)
                          float* __restrict__ hbuf,           // [256 x 256] f32
                          _Float16* __restrict__ ys_f16,      // layer0: [T,256,256] or null
                          float* __restrict__ out_f32,        // layer1: out region or null
                          int t) {
  int wid = blockIdx.x * (blockDim.x >> 5) + (threadIdx.x >> 5);
  if (wid >= 256) return;
  int bt = wid >> 4, ht = wid & 15;
  int lane = threadIdx.x & 31;

  v8f ai = vzero8(), af = vzero8(), ag = vzero8(), ao = vzero8();
  const _Float16* Abase = hin + (size_t)(bt * 16) * 256;
  const _Float16* Bcol = Whh + (size_t)(ht * 16 + (lane & 15)) * 256 + 16 * (lane >> 4);
  for (int k0 = 0; k0 < 256; k0 += 32) {
    v16h a = load_fragA(Abase + k0, 256, lane);
    ai = wmma16(a, load_fragB_ptr(Bcol + k0), ai);
    af = wmma16(a, load_fragB_ptr(Bcol + 256 * 256 + k0), af);
    ag = wmma16(a, load_fragB_ptr(Bcol + 512 * 256 + k0), ag);
    ao = wmma16(a, load_fragB_ptr(Bcol + 768 * 256 + k0), ao);
  }

  int half = lane >> 4, n = lane & 15;
  int j = ht * 16 + n;
  const float* xwt = xw + (size_t)t * 256 * 1024;
#pragma unroll
  for (int r = 0; r < 8; ++r) {
    int row = bt * 16 + half * 8 + r;
    const float* xr = xwt + (size_t)row * 1024 + j;
    float gi = ai[r] + xr[0];
    float gf = af[r] + xr[256];
    float gg = ag[r] + xr[512];
    float go = ao[r] + xr[768];
    float si = fast_sig(gi);
    float sf = fast_sig(gf);
    float so = fast_sig(go);
    float tg = fast_tanh(gg);
    size_t ix = (size_t)row * 256 + j;
    float c = sf * cbuf[ix] + si * tg;
    cbuf[ix] = c;
    float h = so * fast_tanh(c);
    hbuf[ix] = h;
    hout[ix] = (_Float16)h;
    if (ys_f16) ys_f16[(size_t)t * 65536 + ix] = (_Float16)h;
    if (out_f32) {
      int hp = row >> 5, bb = row & 31;
      out_f32[(((size_t)(hp * 65 + t)) * 32 + bb) * 256 + j] = h;
    }
  }
}

// Final gather: h_n/c_n from rows 224..255 of each layer's final h/c.
__global__ void gather_hn(const float* __restrict__ h0, const float* __restrict__ c0,
                          const float* __restrict__ h1, const float* __restrict__ c1,
                          float* __restrict__ out) {
  int idx = blockIdx.x * blockDim.x + threadIdx.x;
  if (idx >= 16384) return;
  int l = idx >> 13; int r = idx & 8191;
  int b = r >> 8; int j = r & 255;
  const float* h = l ? h1 : h0;
  const float* c = l ? c1 : c0;
  size_t src = (size_t)(224 + b) * 256 + j;
  out[idx] = h[src];
  out[16384 + idx] = c[src];
}

// ---------------------------------------------------------------------------
// Host orchestration
// ---------------------------------------------------------------------------
static inline size_t alignup(size_t x) { return (x + 255) & ~(size_t)255; }
static inline int cdiv(int a, int b) { return (a + b - 1) / b; }

enum {
  IN_SRC = 0,
  IN_W1, IN_B1, IN_W2, IN_B2, IN_W3, IN_B3, IN_W4, IN_B4, IN_W5, IN_B5, IN_W6, IN_B6,
  IN_BN1G, IN_BN1B, IN_BN1M, IN_BN1V,
  IN_BN2G, IN_BN2B, IN_BN2M, IN_BN2V,
  IN_BN3G, IN_BN3B, IN_BN3M, IN_BN3V,
  IN_WIH0, IN_WHH0, IN_BIH0, IN_BHH0,
  IN_WIH1, IN_WHH1, IN_BIH1, IN_BHH1,
  IN_POS
};

extern "C" void kernel_launch(void* const* d_in, const int* in_sizes, int n_in,
                              void* d_out, int out_size, void* d_ws, size_t ws_size,
                              hipStream_t stream) {
  (void)in_sizes; (void)n_in; (void)out_size; (void)ws_size;
  char* ws = (char*)d_ws;
  size_t o = 0;
  auto take = [&](size_t bytes) { size_t r = o; o = alignup(o + bytes); return r; };

  // --- workspace layout ---
  size_t W2h = take((size_t)128 * 576 * 2);
  size_t W3h = take((size_t)256 * 1152 * 2);
  size_t W4h = take((size_t)256 * 2304 * 2);
  size_t W5h = take((size_t)512 * 2304 * 2);
  size_t W6h = take((size_t)512 * 4608 * 2);
  size_t Bs2 = take(128 * 4), Bs3 = take(256 * 4), Bs4 = take(256 * 4);
  size_t Bs5 = take(512 * 4), Bs6 = take(512 * 4);
  size_t Wih0h = take((size_t)1024 * 512 * 2);
  size_t Whh0h = take((size_t)1024 * 256 * 2);
  size_t Wih1h = take((size_t)1024 * 256 * 2);
  size_t Whh1h = take((size_t)1024 * 256 * 2);
  size_t Bsum0 = take(1024 * 4), Bsum1 = take(1024 * 4);
  size_t H0a = take(65536 * 2), H0b = take(65536 * 2);
  size_t H1a = take(65536 * 2), H1b = take(65536 * 2);
  size_t C0f = take(65536 * 4), C1f = take(65536 * 4);
  size_t HF0 = take(65536 * 4), HF1 = take(65536 * 4);
  size_t YS0 = take((size_t)65 * 65536 * 2);
  size_t ArA = take((size_t)71860224);   // arena A (max: c2 padded NHWC)
  size_t ArB = take((size_t)68157440);   // arena B (max: xw [16640x1024] f32)

  // NHWC padded activation tensors (1-pixel halo each side)
  _Float16* x0  = (_Float16*)(ws + ArA);  // [32,64,512] (unpadded, Cin=1 path)
  _Float16* p1  = (_Float16*)(ws + ArB);  // [32,34,258,64]
  _Float16* c2  = (_Float16*)(ws + ArA);  // [32,34,258,128]
  _Float16* p2  = (_Float16*)(ws + ArB);  // [32,18,130,128]
  _Float16* c3  = (_Float16*)(ws + ArA);  // [32,18,130,256]
  _Float16* c4  = (_Float16*)(ws + ArB);  // [32,18,130,256]
  _Float16* p4  = (_Float16*)(ws + ArA);  // [32,18,66,256]
  _Float16* c5  = (_Float16*)(ws + ArB);  // [32,18,66,512]
  _Float16* p5  = (_Float16*)(ws + ArA);  // [32,10,66,512]
  _Float16* c6  = (_Float16*)(ws + ArB);  // [32,10,66,512]
  _Float16* seq = (_Float16*)(ws + ArA);  // [65,256,512]
  float*    xw  = (float*)(ws + ArB);     // [16640,1024]

  const int TB = 256;

  // --- weight prep (BN folded into conv3/5/6; K permuted to tap-major) ---
  fold_conv_w<<<cdiv(128 * 576, TB), TB, 0, stream>>>(
      (const float*)d_in[IN_W2], (const float*)d_in[IN_B2], nullptr, nullptr, nullptr, nullptr,
      (_Float16*)(ws + W2h), (float*)(ws + Bs2), 128, 64);
  fold_conv_w<<<cdiv(256 * 1152, TB), TB, 0, stream>>>(
      (const float*)d_in[IN_W3], (const float*)d_in[IN_B3],
      (const float*)d_in[IN_BN1G], (const float*)d_in[IN_BN1B],
      (const float*)d_in[IN_BN1M], (const float*)d_in[IN_BN1V],
      (_Float16*)(ws + W3h), (float*)(ws + Bs3), 256, 128);
  fold_conv_w<<<cdiv(256 * 2304, TB), TB, 0, stream>>>(
      (const float*)d_in[IN_W4], (const float*)d_in[IN_B4], nullptr, nullptr, nullptr, nullptr,
      (_Float16*)(ws + W4h), (float*)(ws + Bs4), 256, 256);
  fold_conv_w<<<cdiv(512 * 2304, TB), TB, 0, stream>>>(
      (const float*)d_in[IN_W5], (const float*)d_in[IN_B5],
      (const float*)d_in[IN_BN2G], (const float*)d_in[IN_BN2B],
      (const float*)d_in[IN_BN2M], (const float*)d_in[IN_BN2V],
      (_Float16*)(ws + W5h), (float*)(ws + Bs5), 512, 256);
  fold_conv_w<<<cdiv(512 * 4608, TB), TB, 0, stream>>>(
      (const float*)d_in[IN_W6], (const float*)d_in[IN_B6],
      (const float*)d_in[IN_BN3G], (const float*)d_in[IN_BN3B],
      (const float*)d_in[IN_BN3M], (const float*)d_in[IN_BN3V],
      (_Float16*)(ws + W6h), (float*)(ws + Bs6), 512, 512);

  cvt_f16<<<cdiv(1024 * 512, TB), TB, 0, stream>>>((const float*)d_in[IN_WIH0], (_Float16*)(ws + Wih0h), 1024 * 512);
  cvt_f16<<<cdiv(1024 * 256, TB), TB, 0, stream>>>((const float*)d_in[IN_WHH0], (_Float16*)(ws + Whh0h), 1024 * 256);
  cvt_f16<<<cdiv(1024 * 256, TB), TB, 0, stream>>>((const float*)d_in[IN_WIH1], (_Float16*)(ws + Wih1h), 1024 * 256);
  cvt_f16<<<cdiv(1024 * 256, TB), TB, 0, stream>>>((const float*)d_in[IN_WHH1], (_Float16*)(ws + Whh1h), 1024 * 256);
  add_bias2<<<cdiv(1024, TB), TB, 0, stream>>>((const float*)d_in[IN_BIH0], (const float*)d_in[IN_BHH0], (float*)(ws + Bsum0), 1024);
  add_bias2<<<cdiv(1024, TB), TB, 0, stream>>>((const float*)d_in[IN_BIH1], (const float*)d_in[IN_BHH1], (float*)(ws + Bsum1), 1024);

  // --- conv stack (NHWC padded) ---
  auto border = [&](_Float16* buf, int Bn, int Hp2, int Wp2, int C) {
    zero_border<<<cdiv(Bn * Hp2 * Wp2 * C, TB), TB, 0, stream>>>(buf, Bn, Hp2, Wp2, C);
  };
  auto conv = [&](const _Float16* in, size_t wh, size_t bias, _Float16* out,
                  int Bn, int Cin, int Hp2, int Wp2, int Cout) {
    int tiles = (Cout >> 5) * ((Bn * (Hp2 - 2) * (Wp2 - 2)) >> 4);
    conv3x3_wmma<<<cdiv(tiles, 4), 128, 0, stream>>>(
        in, (const _Float16*)(ws + wh), (const float*)(ws + bias), out, Bn, Cin, Hp2, Wp2, Cout);
  };
  auto pool = [&](const _Float16* in, _Float16* out, int Bn, int C, int H, int W, int kh, int kw) {
    int tot = Bn * C * (H / kh) * (W / kw);
    maxpool_f16<<<cdiv(tot, TB), TB, 0, stream>>>(in, out, Bn, C, H, W, kh, kw);
  };

  prep_x0<<<cdiv(32 * 64 * 512, TB), TB, 0, stream>>>((const float*)d_in[IN_SRC], x0, 32 * 64 * 512);
  border(p1, 32, 34, 258, 64);
  conv1_pool<<<cdiv(32 * 32 * 256 * 64, TB), TB, 0, stream>>>(
      x0, (const float*)d_in[IN_W1], (const float*)d_in[IN_B1], p1);

  border(c2, 32, 34, 258, 128);
  conv(p1, W2h, Bs2, c2, 32, 64, 34, 258, 128);
  border(p2, 32, 18, 130, 128);
  pool(c2, p2, 32, 128, 32, 256, 2, 2);
  border(c3, 32, 18, 130, 256);
  conv(p2, W3h, Bs3, c3, 32, 128, 18, 130, 256);
  border(c4, 32, 18, 130, 256);
  conv(c3, W4h, Bs4, c4, 32, 256, 18, 130, 256);
  border(p4, 32, 18, 66, 256);
  pool(c4, p4, 32, 256, 16, 128, 1, 2);
  border(c5, 32, 18, 66, 512);
  conv(p4, W5h, Bs5, c5, 32, 256, 18, 66, 512);
  border(p5, 32, 10, 66, 512);
  pool(c5, p5, 32, 512, 16, 64, 2, 1);
  border(c6, 32, 10, 66, 512);
  conv(p5, W6h, Bs6, c6, 32, 512, 10, 66, 512);

  // --- sequence build + LSTM ---
  build_seq<<<cdiv(65 * 256 * 512, TB), TB, 0, stream>>>(c6, (const float*)d_in[IN_POS], seq);

  const int M = 65 * 256;  // 16640
  {
    int tiles = (M >> 4) * (1024 >> 5);
    gemm_xw<<<cdiv(tiles, 4), 128, 0, stream>>>(seq, (const _Float16*)(ws + Wih0h),
                                                (const float*)(ws + Bsum0), xw, M, 1024, 512);
  }
  zero_f16<<<cdiv(65536, TB), TB, 0, stream>>>((_Float16*)(ws + H0a), 65536);
  zero_f32<<<cdiv(65536, TB), TB, 0, stream>>>((float*)(ws + C0f), 65536);
  for (int t = 0; t < 65; ++t) {
    _Float16* hin  = (_Float16*)(ws + ((t & 1) ? H0b : H0a));
    _Float16* hout = (_Float16*)(ws + ((t & 1) ? H0a : H0b));
    lstm_step<<<32, 256, 0, stream>>>(hin, hout, (const _Float16*)(ws + Whh0h), xw,
                                      (float*)(ws + C0f), (float*)(ws + HF0),
                                      (_Float16*)(ws + YS0), nullptr, t);
  }
  {
    int tiles = (M >> 4) * (1024 >> 5);
    gemm_xw<<<cdiv(tiles, 4), 128, 0, stream>>>((const _Float16*)(ws + YS0),
                                                (const _Float16*)(ws + Wih1h),
                                                (const float*)(ws + Bsum1), xw, M, 1024, 256);
  }
  zero_f16<<<cdiv(65536, TB), TB, 0, stream>>>((_Float16*)(ws + H1a), 65536);
  zero_f32<<<cdiv(65536, TB), TB, 0, stream>>>((float*)(ws + C1f), 65536);
  float* out_region = (float*)d_out + 32768;
  for (int t = 0; t < 65; ++t) {
    _Float16* hin  = (_Float16*)(ws + ((t & 1) ? H1b : H1a));
    _Float16* hout = (_Float16*)(ws + ((t & 1) ? H1a : H1b));
    lstm_step<<<32, 256, 0, stream>>>(hin, hout, (const _Float16*)(ws + Whh1h), xw,
                                      (float*)(ws + C1f), (float*)(ws + HF1),
                                      nullptr, out_region, t);
  }

  gather_hn<<<cdiv(16384, TB), TB, 0, stream>>>((const float*)(ws + HF0), (const float*)(ws + C0f),
                                                (const float*)(ws + HF1), (const float*)(ws + C1f),
                                                (float*)d_out);
}
